// TSP_Decoder_18811956756964
// MI455X (gfx1250) — matrix operand order repaired and checked by reading, not compiled, not run on hardware
//
#include <hip/hip_runtime.h>

// ---------------- problem constants ----------------
#define B_ 32
#define P_ 256
#define N_ 1024
#define E_ 512
#define H_ 16
#define D_ 32
#define INV_SQRT_D 0.17677669529663687f
#define INV_SQRT_E 0.044194173824159216f
#define LOGIT_CLIP 10.0f

typedef __attribute__((ext_vector_type(8)))  float  v8f;
typedef __attribute__((ext_vector_type(16))) __bf16 v16bf;
typedef unsigned int v4u __attribute__((ext_vector_type(4)));
typedef int          v8i __attribute__((ext_vector_type(8)));
typedef int          v4i __attribute__((ext_vector_type(4)));

// ---------------- WMMA helper ----------------
static __device__ __forceinline__ v8f wmma_bf(v16bf a, v16bf b, v8f c) {
    // D(f32 16x16) = A(bf16 16x32) * B(bf16 32x16) + C
    return __builtin_amdgcn_wmma_f32_16x16x32_bf16(false, a, false, b,
                                                   (short)0, c, false, false);
}

// A-fragment, 16x32, from f32 row-major rows (row stride = ld floats).
// Layout: lanes 0-15 row=lane, K=0..15; lanes 16-31 row=lane-16, K=16..31
static __device__ __forceinline__ v16bf a_frag_f32(const float* __restrict__ src, int ld) {
    int lane = threadIdx.x & 31, half = lane >> 4, row = lane & 15;
    const float* p = src + (size_t)row * ld + half * 16;
    v16bf a;
#pragma unroll
    for (int i = 0; i < 8; ++i) a[i] = (__bf16)p[i];
#pragma unroll
    for (int i = 0; i < 8; ++i) a[i + 8] = (__bf16)p[i + 8];
    return a;
}
static __device__ __forceinline__ v16bf a_frag_bf16(const __bf16* __restrict__ src, int ld) {
    int lane = threadIdx.x & 31, half = lane >> 4, row = lane & 15;
    const __bf16* p = src + (size_t)row * ld + half * 16;
    v16bf a;
#pragma unroll
    for (int i = 0; i < 8; ++i) a[i] = p[i];
#pragma unroll
    for (int i = 0; i < 8; ++i) a[i + 8] = p[i + 8];
    return a;
}
// B-fragment (32x16) from "transposed" source: src[col][k], col stride ld.
// Layout: lanes 0-15 col=lane K=0..15 ; lanes 16-31 col=lane-16 K=16..31
static __device__ __forceinline__ v16bf bt_frag_f32(const float* __restrict__ src, int ld) {
    int lane = threadIdx.x & 31, half = lane >> 4, col = lane & 15;
    const float* p = src + (size_t)col * ld + half * 16;
    v16bf b;
#pragma unroll
    for (int i = 0; i < 16; ++i) b[i] = (__bf16)p[i];
    return b;
}
// B-fragment from KxN row-major source: src[k][col], row stride ld.
static __device__ __forceinline__ v16bf bkn_frag_f32(const float* __restrict__ src, int ld) {
    int lane = threadIdx.x & 31, half = lane >> 4, col = lane & 15;
    const float* p = src + (size_t)(half * 16) * ld + col;
    v16bf b;
#pragma unroll
    for (int i = 0; i < 16; ++i) b[i] = (__bf16)p[(size_t)i * ld];
    return b;
}
static __device__ __forceinline__ v16bf bkn_frag_bf16(const __bf16* __restrict__ src, int ld) {
    int lane = threadIdx.x & 31, half = lane >> 4, col = lane & 15;
    const __bf16* p = src + (size_t)(half * 16) * ld + col;
    v16bf b;
#pragma unroll
    for (int i = 0; i < 16; ++i) b[i] = p[(size_t)i * ld];
    return b;
}

// reductions within 16-lane halves of the wave (rows of a C tile)
static __device__ __forceinline__ float hmax16(float v) {
    v = fmaxf(v, __shfl_xor(v, 1, 32));
    v = fmaxf(v, __shfl_xor(v, 2, 32));
    v = fmaxf(v, __shfl_xor(v, 4, 32));
    v = fmaxf(v, __shfl_xor(v, 8, 32));
    return v;
}
static __device__ __forceinline__ float hsum16(float v) {
    v += __shfl_xor(v, 1, 32);
    v += __shfl_xor(v, 2, 32);
    v += __shfl_xor(v, 4, 32);
    v += __shfl_xor(v, 8, 32);
    return v;
}

// ---------------- K0: mask pack + transpose: f32 [B,P,N] -> u8 [B,N,P] ----------------
__global__ __launch_bounds__(256) void k_maskT(const float* __restrict__ ninf,
                                               unsigned char* __restrict__ maskT) {
    __shared__ unsigned char t[32][33];
    int n0 = blockIdx.x * 32, p0 = blockIdx.y * 32, b = blockIdx.z;
    int tx = threadIdx.x, ty = threadIdx.y;  // 32 x 8
#pragma unroll
    for (int i = 0; i < 4; ++i) {
        int pl = ty + i * 8;
        float v = ninf[((size_t)b * P_ + p0 + pl) * N_ + n0 + tx];
        t[pl][tx] = (v == 0.0f) ? 1 : 0;
    }
    __syncthreads();
#pragma unroll
    for (int i = 0; i < 4; ++i) {
        int nl = ty + i * 8;
        maskT[((size_t)b * N_ + n0 + nl) * P_ + p0 + tx] = t[tx][nl];
    }
}

// ---------------- K1: q = (enc @ Wq^T + q_first) * 1/sqrt(D) -> bf16 [B,H,P,D] --------
__global__ __launch_bounds__(256) void k_qgemm(const float* __restrict__ enc,
                                               const float* __restrict__ Wq,
                                               const float* __restrict__ qfirst,
                                               __bf16* __restrict__ qs) {
    int wv = threadIdx.x >> 5, lane = threadIdx.x & 31;
    int half = lane >> 4, tc = lane & 15;
    int mt = wv & 3, ch = wv >> 2;
    int row0 = blockIdx.x * 64 + mt * 16;  // global row in [0, B*P)
    int j0 = ch * 256;
    v8f acc[16];
    v8f z = {};
#pragma unroll
    for (int t = 0; t < 16; ++t) acc[t] = z;
    for (int e0 = 0; e0 < E_; e0 += 32) {
        v16bf va = a_frag_f32(enc + (size_t)row0 * E_ + e0, E_);
#pragma unroll
        for (int ct = 0; ct < 16; ++ct) {
            v16bf vb = bt_frag_f32(Wq + (size_t)(j0 + ct * 16) * E_ + e0, E_);
            acc[ct] = wmma_bf(va, vb, acc[ct]);
        }
    }
#pragma unroll
    for (int ct = 0; ct < 16; ++ct) {
#pragma unroll
        for (int r = 0; r < 8; ++r) {
            int i = row0 + r + 8 * half;
            int j = j0 + ct * 16 + tc;
            int b = i >> 8, p = i & 255, h = j >> 5, d = j & 31;
            size_t qi = (((size_t)b * H_ + h) * P_ + p) * D_ + d;
            float q = acc[ct][r] + qfirst[qi];
            qs[qi] = (__bf16)(q * INV_SQRT_D);
        }
    }
}

// ---------------- K2: flash attention per (b,h) -> out_concat bf16 [B,P,H*D] ----------
__global__ __launch_bounds__(256) void k_attn(const __bf16* __restrict__ qs,
                                              const float* __restrict__ kg,
                                              const float* __restrict__ vg,
                                              const unsigned char* __restrict__ maskT,
                                              __bf16* __restrict__ oc) {
    __shared__ __bf16 ldsP[8][16 * 32];  // per-wave C->A transpose buffer
    int bh = blockIdx.x, b = bh >> 4, h = bh & 15;
    int wv = threadIdx.x >> 5, lane = threadIdx.x & 31;
    int half = lane >> 4, tc = lane & 15;
    int p0 = wv * 32;
    const __bf16* q_bh = qs + (size_t)bh * P_ * D_;
    const float* k_bh = kg + (size_t)bh * N_ * D_;
    const float* v_bh = vg + (size_t)bh * N_ * D_;

    v16bf qf[2];
    qf[0] = a_frag_bf16(q_bh + (size_t)p0 * D_, D_);
    qf[1] = a_frag_bf16(q_bh + (size_t)(p0 + 16) * D_, D_);

    float m[2][8], l[2][8];
    v8f o[2][2];
    v8f z = {};
#pragma unroll
    for (int mt = 0; mt < 2; ++mt) {
        o[mt][0] = z; o[mt][1] = z;
#pragma unroll
        for (int r = 0; r < 8; ++r) { m[mt][r] = -1e30f; l[mt][r] = 0.0f; }
    }

    for (int n0 = 0; n0 < N_; n0 += 32) {
        if (n0 + 32 < N_) __builtin_prefetch(k_bh + (size_t)(n0 + 32) * D_, 0, 1);
        v16bf kflo = bt_frag_f32(k_bh + (size_t)n0 * D_, D_);
        v16bf kfhi = bt_frag_f32(k_bh + (size_t)(n0 + 16) * D_, D_);
        v16bf vflo = bkn_frag_f32(v_bh + (size_t)n0 * D_ + 0, D_);
        v16bf vfhi = bkn_frag_f32(v_bh + (size_t)n0 * D_ + 16, D_);
#pragma unroll
        for (int mt = 0; mt < 2; ++mt) {
            v8f s0 = wmma_bf(qf[mt], kflo, z);
            v8f s1 = wmma_bf(qf[mt], kfhi, z);
            unsigned long long mk0 = *(const unsigned long long*)(maskT +
                ((size_t)b * N_ + n0 + tc) * P_ + p0 + mt * 16 + half * 8);
            unsigned long long mk1 = *(const unsigned long long*)(maskT +
                ((size_t)b * N_ + n0 + 16 + tc) * P_ + p0 + mt * 16 + half * 8);
#pragma unroll
            for (int r = 0; r < 8; ++r) {
                float tmax = hmax16(fmaxf(s0[r], s1[r]));
                float nm = fmaxf(m[mt][r], tmax);
                float sc = __expf(m[mt][r] - nm);
                m[mt][r] = nm;
                float pa = __expf(s0[r] - nm) * (float)((mk0 >> (8 * r)) & 1ull);
                float pb = __expf(s1[r] - nm) * (float)((mk1 >> (8 * r)) & 1ull);
                l[mt][r] = l[mt][r] * sc + pa + pb;
                o[mt][0][r] *= sc;
                o[mt][1][r] *= sc;
                int row = r + 8 * half;
                ldsP[wv][row * 32 + tc] = (__bf16)pa;
                ldsP[wv][row * 32 + 16 + tc] = (__bf16)pb;
            }
            v16bf pfr = a_frag_bf16(&ldsP[wv][0], 32);  // per-wave LDS, in-order RAW
            o[mt][0] = wmma_bf(pfr, vflo, o[mt][0]);
            o[mt][1] = wmma_bf(pfr, vfhi, o[mt][1]);
        }
    }
#pragma unroll
    for (int mt = 0; mt < 2; ++mt) {
#pragma unroll
        for (int r = 0; r < 8; ++r) {
            float inv = 1.0f / hsum16(l[mt][r]);
            int p = p0 + mt * 16 + r + 8 * half;
#pragma unroll
            for (int dt = 0; dt < 2; ++dt) {
                size_t idx = ((size_t)b * P_ + p) * (H_ * D_) + h * D_ + dt * 16 + tc;
                oc[idx] = (__bf16)(o[mt][dt][r] * inv);
            }
        }
    }
}

// ---------------- K3: mh = out_concat @ W_combine^T -> bf16 [B,P,E] -------------------
__global__ __launch_bounds__(256) void k_cgemm(const __bf16* __restrict__ oc,
                                               const float* __restrict__ Wc,
                                               __bf16* __restrict__ mh) {
    int wv = threadIdx.x >> 5, lane = threadIdx.x & 31;
    int half = lane >> 4, tc = lane & 15;
    int mt = wv & 3, ch = wv >> 2;
    int row0 = blockIdx.x * 64 + mt * 16;
    int j0 = ch * 256;
    v8f acc[16];
    v8f z = {};
#pragma unroll
    for (int t = 0; t < 16; ++t) acc[t] = z;
    for (int e0 = 0; e0 < E_; e0 += 32) {
        v16bf va = a_frag_bf16(oc + (size_t)row0 * E_ + e0, E_);
#pragma unroll
        for (int ct = 0; ct < 16; ++ct) {
            v16bf vb = bt_frag_f32(Wc + (size_t)(j0 + ct * 16) * E_ + e0, E_);
            acc[ct] = wmma_bf(va, vb, acc[ct]);
        }
    }
#pragma unroll
    for (int ct = 0; ct < 16; ++ct) {
#pragma unroll
        for (int r = 0; r < 8; ++r) {
            int i = row0 + r + 8 * half;
            int j = j0 + ct * 16 + tc;
            mh[(size_t)i * E_ + j] = (__bf16)acc[ct][r];
        }
    }
}

// ------- K4: e = exp(10*tanh((mh@shk)/sqrtE)) * mask ; rowsum -> rs ; e -> d_out ------
// The WG's 128 mh rows (contiguous 128KB bf16) are brought into LDS once with the
// Tensor Data Mover (2D tile: 512 elem x 128 rows, data_size=2B), tracked by TENSORcnt.
__global__ __launch_bounds__(256) void k_pointer(const __bf16* __restrict__ mh,
                                                 const float* __restrict__ shk,
                                                 const unsigned char* __restrict__ maskT,
                                                 float* __restrict__ out,
                                                 float* __restrict__ rs) {
    extern __shared__ char smem[];
    __bf16* sm_mh = (__bf16*)smem;              // 128 x 512 bf16 = 128KB
    __bf16* sk    = (__bf16*)(smem + 131072);   // 32 x 128 bf16 = 8KB
    int b = blockIdx.x >> 1, pt = blockIdx.x & 1;
    int wv = threadIdx.x >> 5, lane = threadIdx.x & 31;
    int half = lane >> 4, tc = lane & 15;
    int prow0 = pt * 128 + wv * 16;
    const float* shk_b = shk + (size_t)b * E_ * N_;
    const __bf16* mh_rows = mh + ((size_t)b * P_ + pt * 128) * E_;

    if (wv == 0) {  // one wave issues the TDM load of the 128x512 bf16 mh tile
        unsigned long long va = (unsigned long long)(uintptr_t)mh_rows;
        unsigned int ldsa = (unsigned int)(uintptr_t)sm_mh;  // flat low 32b = LDS byte addr
        v4u g0 = { 1u,                                        // count=1 valid descriptor
                   ldsa,                                      // lds_addr [63:32]
                   (unsigned int)(va & 0xFFFFFFFFull),        // global_addr [95:64]
                   (unsigned int)((va >> 32) & 0x01FFFFFFull) // global_addr [120:96]
                       | (2u << 30) };                        // type=2 ("image")
        v8i g1 = { (int)(1u << 16),    // data_size=1 (2 bytes)
                   (int)(512u << 16),  // tensor_dim0 = 512 (low16 @ bit48)
                   (int)(128u << 16),  // tensor_dim0 hi=0 | tensor_dim1=128 (@ bit80)
                   (int)(512u << 16),  // tensor_dim1 hi=0 | tile_dim0=512 (@ bit112)
                   (int)128,           // tile_dim1=128, tile_dim2=0
                   (int)512,           // tensor_dim0_stride = 512 (low32 @ bit160)
                   0, 0 };
        v4i z4 = { 0, 0, 0, 0 };
#if __clang_major__ >= 23
        v8i z8 = { 0, 0, 0, 0, 0, 0, 0, 0 };
        __builtin_amdgcn_tensor_load_to_lds(g0, g1, z4, z4, z8, 0);
#else
        __builtin_amdgcn_tensor_load_to_lds(g0, g1, z4, z4, 0);
#endif
        __builtin_amdgcn_s_wait_tensorcnt(0);
    }
    __syncthreads();

    float se[8];
#pragma unroll
    for (int r = 0; r < 8; ++r) se[r] = 0.0f;
    v8f z = {};

    for (int cb = 0; cb < 8; ++cb) {
        int c0 = cb * 128;
        v8f acc[8];
#pragma unroll
        for (int t = 0; t < 8; ++t) acc[t] = z;
        for (int e0 = 0; e0 < E_; e0 += 32) {
            __syncthreads();
#pragma unroll
            for (int it = 0; it < 4; ++it) {  // stage 32x128 f32 -> bf16 LDS
                int idx = threadIdx.x + 256 * it;  // over [32][32 x float4]
                int e = idx >> 5, nc = (idx & 31) * 4;
                float4 f = *(const float4*)(shk_b + (size_t)(e0 + e) * N_ + c0 + nc);
                __bf16* d = sk + e * 128 + nc;
                d[0] = (__bf16)f.x; d[1] = (__bf16)f.y;
                d[2] = (__bf16)f.z; d[3] = (__bf16)f.w;
            }
            __syncthreads();
            v16bf va = a_frag_bf16(sm_mh + (size_t)(wv * 16) * E_ + e0, E_);
#pragma unroll
            for (int ct = 0; ct < 8; ++ct) {
                v16bf vb = bkn_frag_bf16(sk + ct * 16, 128);
                acc[ct] = wmma_bf(va, vb, acc[ct]);
            }
        }
#pragma unroll
        for (int ct = 0; ct < 8; ++ct) {
            int n = c0 + ct * 16 + tc;
            unsigned long long mk = *(const unsigned long long*)(maskT +
                ((size_t)b * N_ + n) * P_ + prow0 + half * 8);
#pragma unroll
            for (int r = 0; r < 8; ++r) {
                float s = acc[ct][r] * INV_SQRT_E;
                float cl = LOGIT_CLIP * tanhf(s);
                float e = __expf(cl) * (float)((mk >> (8 * r)) & 1ull);
                out[((size_t)b * P_ + prow0 + half * 8 + r) * N_ + n] = e;
                se[r] += e;
            }
        }
    }
#pragma unroll
    for (int r = 0; r < 8; ++r) se[r] = hsum16(se[r]);
    if (tc == 0) {  // lanes 0 (rows r) and 16 (rows r+8)
#pragma unroll
        for (int r = 0; r < 8; ++r)
            rs[(size_t)b * P_ + prow0 + half * 8 + r] = se[r];
    }
}

// ---------------- K5: in-place normalize: out[b,p,n] /= rs[b,p] -----------------------
__global__ __launch_bounds__(256) void k_norm(float* __restrict__ out,
                                              const float* __restrict__ rs) {
    size_t gid = (size_t)blockIdx.x * 256 + threadIdx.x;
    size_t idx = gid * 4;
    float inv = 1.0f / rs[idx >> 10];  // row = idx / N
    float4 v = *(float4*)(out + idx);
    v.x *= inv; v.y *= inv; v.z *= inv; v.w *= inv;
    *(float4*)(out + idx) = v;
}

// ---------------- launcher ----------------
extern "C" void kernel_launch(void* const* d_in, const int* in_sizes, int n_in,
                              void* d_out, int out_size, void* d_ws, size_t ws_size,
                              hipStream_t stream) {
    const float* enc    = (const float*)d_in[0];  // [B,P,E]
    const float* ninf   = (const float*)d_in[1];  // [B,P,N]
    const float* qfirst = (const float*)d_in[2];  // [B,H,P,D]
    const float* kmat   = (const float*)d_in[3];  // [B,H,N,D]
    const float* vmat   = (const float*)d_in[4];  // [B,H,N,D]
    const float* shk    = (const float*)d_in[5];  // [B,E,N]
    const float* Wq     = (const float*)d_in[6];  // [E,E]
    const float* Wc     = (const float*)d_in[7];  // [E,E]
    float* out = (float*)d_out;                   // [B,P,N]

    char* ws = (char*)d_ws;
    __bf16* qs = (__bf16*)(ws + 0);                       // [B,H,P,D]  8 MB
    __bf16* oc = (__bf16*)(ws + (8u << 20));              // [B,P,E]    8 MB
    __bf16* mh = (__bf16*)(ws + (16u << 20));             // [B,P,E]    8 MB
    unsigned char* maskT = (unsigned char*)(ws + (24u << 20));  // [B,N,P] 8 MB
    float* rsum = (float*)(ws + (32u << 20));             // [B,P]     32 KB

    k_maskT<<<dim3(N_ / 32, P_ / 32, B_), dim3(32, 8, 1), 0, stream>>>(ninf, maskT);
    k_qgemm<<<dim3((B_ * P_) / 64), dim3(256), 0, stream>>>(enc, Wq, qfirst, qs);
    k_attn<<<dim3(B_ * H_), dim3(256), 0, stream>>>(qs, kmat, vmat, maskT, oc);
    k_cgemm<<<dim3((B_ * P_) / 64), dim3(256), 0, stream>>>(oc, Wc, mh);
    k_pointer<<<dim3(B_ * 2), dim3(256), 139264, stream>>>(mh, shk, maskT, out, rsum);
    k_norm<<<dim3((B_ * P_ * N_ / 4) / 256), dim3(256), 0, stream>>>(out, rsum);
}
